// RelationalLayer_69990787055720
// MI455X (gfx1250) — matrix operand order repaired and checked by reading, not compiled, not run on hardware
//
#include <hip/hip_runtime.h>

typedef __attribute__((ext_vector_type(16))) _Float16 v16h;
typedef __attribute__((ext_vector_type(8)))  _Float16 v8h;
typedef __attribute__((ext_vector_type(8)))  float    v8f;

#define BDIM 16
#define NDIM 1024
#define DDIM 64
#define KDIM 8

// ---------------------------------------------------------------------------
// Stage 1: z[k,b,n,e] = relu( sum_d x[b,n,d] * W[k,d,e] + bias[k,e] ), f16 out
// One block per (k,b,n) row; 64 threads = one output element each.
// x row broadcast through LDS; W reads coalesced across e.
// ---------------------------------------------------------------------------
__global__ __launch_bounds__(64) void stage1_proj_relu(
    const float* __restrict__ x, const float* __restrict__ W,
    const float* __restrict__ bias, _Float16* __restrict__ z)
{
    __shared__ float xs[DDIM];
    const int blk = blockIdx.x;           // k*B*N + b*N + n
    const int n  = blk & (NDIM - 1);
    const int kb = blk >> 10;
    const int b  = kb & (BDIM - 1);
    const int k  = kb >> 4;
    const int e  = threadIdx.x;

    xs[e] = x[(size_t)(b * NDIM + n) * DDIM + e];
    __syncthreads();

    const float* Wk = W + (size_t)k * DDIM * DDIM;
    float acc = bias[k * DDIM + e];
#pragma unroll 8
    for (int d = 0; d < DDIM; ++d)
        acc = fmaf(xs[d], Wk[d * DDIM + e], acc);
    acc = acc > 0.0f ? acc : 0.0f;

    z[((size_t)(k * BDIM + b) * NDIM + n) * DDIM + e] = (_Float16)acc;
}

// ---------------------------------------------------------------------------
// Stage 2: R[b,n,m,k] = sum_e z[k,b,n,e] * z[k,b,m,e]
// Block = 256 threads = 8 waves; wave w handles k = w.
// Each wave: 32x32 (n,m) tile = 2x2 fragments of 16x16, inner dim 64 split
// into two k=32 chunks -> 8x v_wmma_f32_16x16x32_f16 per wave.
// Gram structure => B operand (z^T) fragments are contiguous z-row chunks:
// direct global b64/b128 loads, no LDS, no transpose instructions needed.
// ---------------------------------------------------------------------------
__device__ __forceinline__ v16h cat8(v8h lo, v8h hi) {
    return __builtin_shufflevector(lo, hi,
        0, 1, 2, 3, 4, 5, 6, 7, 8, 9, 10, 11, 12, 13, 14, 15);
}

__global__ __launch_bounds__(256) void stage2_gram(
    const _Float16* __restrict__ z, float* __restrict__ R)
{
    const int b    = blockIdx.y;
    const int nt   = blockIdx.x >> 5;          // 32 n-tiles
    const int mt   = blockIdx.x & 31;          // 32 m-tiles
    const int n0   = nt << 5;
    const int m0   = mt << 5;
    const int k    = threadIdx.x >> 5;         // wave id == attribute k
    const int lane = threadIdx.x & 31;
    const int half = lane >> 4;                // lane group 0/1
    const int l16  = lane & 15;

    const _Float16* zkb = z + (size_t)(k * BDIM + b) * NDIM * DDIM;

    v8f acc00 = {}, acc01 = {}, acc10 = {}, acc11 = {};

#pragma unroll
    for (int c = 0; c < 2; ++c) {
        const int e0 = c << 5;                 // k-chunk base (0, 32)

        // A fragments (16x32 f16): lane holds row n0+16r+l16,
        // halfs[0..7] = e0+half*8.., halfs[8..15] = e0+16+half*8..
        const _Float16* ar0 = zkb + (size_t)(n0 + l16)      * DDIM + e0 + half * 8;
        const _Float16* ar1 = zkb + (size_t)(n0 + 16 + l16) * DDIM + e0 + half * 8;
        v16h a0 = cat8(*(const v8h*)ar0, *(const v8h*)(ar0 + 16));
        v16h a1 = cat8(*(const v8h*)ar1, *(const v8h*)(ar1 + 16));

        // B fragments (32x16 f16, B = z^T): lane holds col m0+16s+l16,
        // halfs[0..7] = e0+half*8.., halfs[8..15] = e0+16+half*8.. (same as A).
        const _Float16* br0 = zkb + (size_t)(m0 + l16)      * DDIM + e0 + half * 8;
        const _Float16* br1 = zkb + (size_t)(m0 + 16 + l16) * DDIM + e0 + half * 8;
        v16h b0 = cat8(*(const v8h*)br0, *(const v8h*)(br0 + 16));
        v16h b1 = cat8(*(const v8h*)br1, *(const v8h*)(br1 + 16));

        acc00 = __builtin_amdgcn_wmma_f32_16x16x32_f16(false, a0, false, b0,
                                                       (short)0, acc00, false, false);
        acc01 = __builtin_amdgcn_wmma_f32_16x16x32_f16(false, a0, false, b1,
                                                       (short)0, acc01, false, false);
        acc10 = __builtin_amdgcn_wmma_f32_16x16x32_f16(false, a1, false, b0,
                                                       (short)0, acc10, false, false);
        acc11 = __builtin_amdgcn_wmma_f32_16x16x32_f16(false, a1, false, b1,
                                                       (short)0, acc11, false, false);
    }

    // C/D layout: VGPR j, lanes 0-15 -> M=j, lanes 16-31 -> M=8+j; col = lane&15.
    // Output R[b][n][m][k]; write-once 512MB -> nontemporal stores.
    float* Rb = R + (size_t)b * NDIM * NDIM * KDIM;
    const int col0 = m0 + l16;
    const int col1 = m0 + 16 + l16;
#pragma unroll
    for (int j = 0; j < 8; ++j) {
        const int rn0 = n0 + 8 * half + j;
        const int rn1 = rn0 + 16;
        __builtin_nontemporal_store(acc00[j], &Rb[((size_t)rn0 * NDIM + col0) * KDIM + k]);
        __builtin_nontemporal_store(acc01[j], &Rb[((size_t)rn0 * NDIM + col1) * KDIM + k]);
        __builtin_nontemporal_store(acc10[j], &Rb[((size_t)rn1 * NDIM + col0) * KDIM + k]);
        __builtin_nontemporal_store(acc11[j], &Rb[((size_t)rn1 * NDIM + col1) * KDIM + k]);
    }
}

extern "C" void kernel_launch(void* const* d_in, const int* in_sizes, int n_in,
                              void* d_out, int out_size, void* d_ws, size_t ws_size,
                              hipStream_t stream) {
    const float* x    = (const float*)d_in[0];   // [B, N, D_IN]
    const float* W    = (const float*)d_in[1];   // [K, D_IN, D_PROJ]
    const float* bias = (const float*)d_in[2];   // [K, D_PROJ]
    float*       out  = (float*)d_out;           // [B, N, N, K]
    _Float16*    zws  = (_Float16*)d_ws;         // [K, B, N, D_PROJ] f16, 16 MB

    // Stage 1: projection + relu into f16 workspace
    stage1_proj_relu<<<dim3(KDIM * BDIM * NDIM), dim3(64), 0, stream>>>(x, W, bias, zws);

    // Stage 2: per-(b,k) Gram matrices via WMMA; grid = (n-tiles * m-tiles, B)
    stage2_gram<<<dim3((NDIM / 32) * (NDIM / 32), BDIM), dim3(256), 0, stream>>>(zws, out);
}
